// RGLRU_74594991997379
// MI455X (gfx1250) — compile-verified, matrix operand order checked by reading, not run on hardware
//
#include <hip/hip_runtime.h>
#include <hip/hip_bf16.h>
#include <cstdint>

#define B_   4
#define L_   4096
#define W_   2560
#define H_   10
#define BW_  256
#define KP_  8            // K panels of 32 (BW_/32)
#define TPW_ 4            // row tiles per wave
#define WPB_ 8            // waves per block

typedef __attribute__((ext_vector_type(16))) __bf16 v16bf;
typedef __attribute__((ext_vector_type(8)))  float  v8f;

__device__ __forceinline__ float sigmoidf_(float z) {
    return 1.0f / (1.0f + __expf(-z));
}

// ---------------------------------------------------------------------------
// Kernel 1: block-diagonal gate matmuls (bf16 WMMA) + gating epilogue.
//   grid  = (1024/(WPB_*TPW_)=32, H_*16=160)
//   block = 256 (8 waves); each wave owns one 16-col n-tile of one head and
//   iterates TPW_ row tiles with the weight panels resident in VGPRs.
// ---------------------------------------------------------------------------
__global__ __launch_bounds__(256)
void rglru_gates_wmma(const float* __restrict__ x,
                      const int*   __restrict__ segment_pos,
                      const float* __restrict__ a_param,
                      const float* __restrict__ w_x,
                      const float* __restrict__ b_x,
                      const float* __restrict__ w_a,
                      const float* __restrict__ b_a,
                      float* __restrict__ xn_out,   // d_out (y region), overwritten by scan
                      float* __restrict__ a_out)    // d_ws: masked decay a
{
    const int lane = threadIdx.x & 31;
    const int wave = threadIdx.x >> 5;
    const int hn   = blockIdx.y;          // 0..159  = h * 16 + n_tile
    const int h    = hn >> 4;
    const int n0   = (hn & 15) << 4;
    const int nlo  = lane & 15;
    const int hi   = lane >> 4;           // lane half
    const int koff = hi << 3;             // 0 or 8 (ISA 16-bit A/B lane layout)

    // ---- Load both gates' weight tiles (B operands) into registers as bf16.
    // B is used as (B^T) in the A-style layout: lane -> column n, element e ->
    //   e<8 : K = 32p + koff + e ; e>=8 : K = 32p + 16 + koff + (e-8)
    v16bf bx[KP_], ba[KP_];
    {
        const size_t wbase = (size_t)h * BW_ * BW_ + (size_t)(n0 + nlo);
        #pragma unroll
        for (int p = 0; p < KP_; ++p) {
            #pragma unroll
            for (int e = 0; e < 16; ++e) {
                const int K = 32 * p + koff + (e < 8 ? e : 16 + (e - 8));
                bx[p][e] = (__bf16)w_x[wbase + (size_t)K * BW_];
                ba[p][e] = (__bf16)w_a[wbase + (size_t)K * BW_];
            }
        }
    }

    const int   col  = h * BW_ + n0 + nlo;               // channel index in W
    const float bxv  = b_x[h * BW_ + n0 + nlo];
    const float bav  = b_a[h * BW_ + n0 + nlo];
    const float c8   = 8.0f * log1pf(__expf(a_param[col]));   // 8*softplus(a_param)

    #pragma unroll 1
    for (int i = 0; i < TPW_; ++i) {
        const int mt   = (blockIdx.x * WPB_ + wave) * TPW_ + i;   // row tile 0..1023
        const int row0 = mt * 16;
        const int arow = row0 + nlo;                              // A row this lane feeds
        const float* xr = x + (size_t)arow * W_ + h * BW_;

        v8f accx = {}; v8f acca = {};
        #pragma unroll
        for (int p = 0; p < KP_; ++p) {
            // Two contiguous 8-float runs per lane per panel -> 4x b128 loads.
            const float4 f0 = *(const float4*)(xr + 32 * p + koff);
            const float4 f1 = *(const float4*)(xr + 32 * p + koff + 4);
            const float4 f2 = *(const float4*)(xr + 32 * p + koff + 16);
            const float4 f3 = *(const float4*)(xr + 32 * p + koff + 20);
            v16bf At;
            At[0]  = (__bf16)f0.x; At[1]  = (__bf16)f0.y; At[2]  = (__bf16)f0.z; At[3]  = (__bf16)f0.w;
            At[4]  = (__bf16)f1.x; At[5]  = (__bf16)f1.y; At[6]  = (__bf16)f1.z; At[7]  = (__bf16)f1.w;
            At[8]  = (__bf16)f2.x; At[9]  = (__bf16)f2.y; At[10] = (__bf16)f2.z; At[11] = (__bf16)f2.w;
            At[12] = (__bf16)f3.x; At[13] = (__bf16)f3.y; At[14] = (__bf16)f3.z; At[15] = (__bf16)f3.w;

            accx = __builtin_amdgcn_wmma_f32_16x16x32_bf16(
                       false, At, false, bx[p], (short)0, accx, false, false);
            acca = __builtin_amdgcn_wmma_f32_16x16x32_bf16(
                       false, At, false, ba[p], (short)0, acca, false, false);
        }

        // Epilogue: C/D layout -> VGPR r holds (M = r + 8*hi, N = nlo).
        #pragma unroll
        for (int r = 0; r < 8; ++r) {
            const int    row = row0 + r + (hi << 3);      // global row in [0, B*L)
            const size_t off = (size_t)row * W_ + col;
            const bool   rst = (segment_pos[row] == 0);

            const float gx   = sigmoidf_(accx[r] + bxv);
            const float ga   = sigmoidf_(acca[r] + bav);
            const float av   = __expf(-c8 * ga);
            const float mult = rst ? 1.0f : sqrtf(fmaxf(0.0f, 1.0f - av * av));
            const float xv   = x[off];

            xn_out[off] = xv * gx * mult;                 // normalized_x
            a_out[off]  = rst ? 0.0f : av;                // reset-masked decay
        }
    }
}

// ---------------------------------------------------------------------------
// Kernel 2: diagonal linear recurrence h_t = a_t*h_{t-1} + x_t, per channel.
// One thread per (b, w); coalesced across w. y computed in place over xn.
// ---------------------------------------------------------------------------
__global__ __launch_bounds__(256)
void rglru_scan(const float* __restrict__ a,
                float* __restrict__ y,        // in: normalized_x, out: y
                float* __restrict__ y_last)
{
    const int idx = blockIdx.x * blockDim.x + threadIdx.x;
    if (idx >= B_ * W_) return;
    const int b = idx / W_;
    const int w = idx - b * W_;
    const size_t base = (size_t)b * L_ * W_ + w;

    float h = 0.0f;
    #pragma unroll 4
    for (int l = 0; l < L_; ++l) {
        const size_t o = base + (size_t)l * W_;
        h = fmaf(a[o], h, y[o]);
        y[o] = h;
    }
    y_last[idx] = h;
}

// ---------------------------------------------------------------------------
extern "C" void kernel_launch(void* const* d_in, const int* in_sizes, int n_in,
                              void* d_out, int out_size, void* d_ws, size_t ws_size,
                              hipStream_t stream) {
    const float* x        = (const float*)d_in[0];
    const int*   seg_pos  = (const int*)  d_in[1];
    const float* a_param  = (const float*)d_in[2];
    const float* w_x      = (const float*)d_in[3];
    const float* b_x      = (const float*)d_in[4];
    const float* w_a      = (const float*)d_in[5];
    const float* b_a      = (const float*)d_in[6];

    float* y      = (float*)d_out;                       // [B, L, W]
    float* y_last = y + (size_t)B_ * L_ * W_;            // [B, W]
    float* a_ws   = (float*)d_ws;                        // [B, L, W] decay scratch

    // Kernel 1: 1024 row tiles / (8 waves * 4 tiles) = 32 blocks in x;
    //           H*16 (head, n-tile) pairs in y.
    dim3 g1(32, H_ * 16, 1);
    rglru_gates_wmma<<<g1, 256, 0, stream>>>(x, seg_pos, a_param,
                                             w_x, b_x, w_a, b_a,
                                             y, a_ws);

    // Kernel 2: one thread per channel (B*W = 10240).
    const int nch = B_ * W_;
    rglru_scan<<<(nch + 255) / 256, 256, 0, stream>>>(a_ws, y, y_last);
}
